// CLIP_Cross_Attention_VS_38328288150359
// MI455X (gfx1250) — matrix operand order repaired and checked by reading, not compiled, not run on hardware
//
#include <hip/hip_runtime.h>
#include <hip/hip_bf16.h>

// ---------------------------------------------------------------------------
// CLIP cross-attention (text->vision), MI455X / gfx1250.
// bf16 WMMA pipeline + TDM (tensor_load_to_lds) double-buffered GEMM B tiles.
// ---------------------------------------------------------------------------

#ifndef __has_builtin
#define __has_builtin(x) 0
#endif

#if __has_builtin(__builtin_amdgcn_tensor_load_to_lds)
#define HAVE_TDM 1
#if __has_include(<hip/amd_detail/amd_gfx1250_TDM.h>)
#define TDM_ARITY6 1   // therock-10.0 headers => 6-arg builtin
#endif
#endif

typedef __attribute__((ext_vector_type(16))) __bf16        v16bf;
typedef __attribute__((ext_vector_type(8)))  float         v8f;
typedef __attribute__((ext_vector_type(4)))  float         f32x4;
typedef __attribute__((ext_vector_type(8)))  unsigned int  u32x8;
typedef __attribute__((ext_vector_type(4)))  unsigned int  u32x4;
typedef __attribute__((ext_vector_type(4)))  unsigned int  tdm_u32x4;
typedef __attribute__((ext_vector_type(8)))  int           tdm_i32x8;
typedef __attribute__((ext_vector_type(4)))  int           tdm_i32x4;

#define NB     16      // batch
#define LT     128     // text length
#define LV     4096    // vision length
#define DM     1024    // model dim
#define NH     16      // heads
#define HD     64      // head dim

// ----------------------------- helpers -------------------------------------

__device__ __forceinline__ unsigned short f2bf(float f) {
  unsigned u = __builtin_bit_cast(unsigned, f);
  u = (u + 0x7FFFu + ((u >> 16) & 1u)) >> 16;     // round-to-nearest-even
  return (unsigned short)u;
}

__device__ __forceinline__ unsigned pk2bf(float lo, float hi) {
  unsigned a = __builtin_bit_cast(unsigned, lo);
  unsigned b = __builtin_bit_cast(unsigned, hi);
  a = (a + 0x7FFFu + ((a >> 16) & 1u)) >> 16;
  b = (b + 0x7FFFu + ((b >> 16) & 1u)) & 0xFFFF0000u;
  return (a & 0xFFFFu) | b;
}

__device__ __forceinline__ v8f vzero8() {
  v8f z = {0.f, 0.f, 0.f, 0.f, 0.f, 0.f, 0.f, 0.f};
  return z;
}

__device__ __forceinline__ v8f wmma_bf16(u32x8 a, u32x8 b, v8f c) {
  return __builtin_amdgcn_wmma_f32_16x16x32_bf16(
      false, __builtin_bit_cast(v16bf, a),
      false, __builtin_bit_cast(v16bf, b),
      (short)0, c, false, false);
}

// B-fragment (32x16, bf16): lane L holds col N=L%16; e0..7 -> K = kb..kb+7,
// e8..15 -> K = kb+16..kb+23 from a K-contiguous row; caller adds the
// +8-halves lane offset (kb = L<16 ? 0 : 8).
__device__ __forceinline__ u32x8 load_b16(const unsigned short* p) {
  const u32x4* q = (const u32x4*)p;
  u32x4 x = q[0], y = q[2];
  u32x8 r;
  r[0] = x[0]; r[1] = x[1]; r[2] = x[2]; r[3] = x[3];
  r[4] = y[0]; r[5] = y[1]; r[6] = y[2]; r[7] = y[3];
  return r;
}

// A-fragment (16x32, bf16): lane L row M=L%16; halves e0..7 -> K = kb..kb+7,
// e8..15 -> K = kb+16..kb+23, where kb = k0 + (L>=16 ? 8 : 0).
__device__ __forceinline__ u32x8 load_a_bf16(const unsigned short* A, long row,
                                             int ld, int kb) {
  const unsigned short* p = A + row * (long)ld + kb;
  u32x4 x = *(const u32x4*)p;
  u32x4 y = *(const u32x4*)(p + 16);
  u32x8 r;
  r[0] = x[0]; r[1] = x[1]; r[2] = x[2]; r[3] = x[3];
  r[4] = y[0]; r[5] = y[1]; r[6] = y[2]; r[7] = y[3];
  return r;
}

__device__ __forceinline__ u32x8 load_a_f32(const float* A, long row, int ld,
                                            int kb) {
  const float* p = A + row * (long)ld + kb;
  f32x4 x0 = ((const f32x4*)p)[0];
  f32x4 x1 = ((const f32x4*)p)[1];
  f32x4 y0 = ((const f32x4*)(p + 16))[0];
  f32x4 y1 = ((const f32x4*)(p + 16))[1];
  u32x8 r;
  r[0] = pk2bf(x0[0], x0[1]); r[1] = pk2bf(x0[2], x0[3]);
  r[2] = pk2bf(x1[0], x1[1]); r[3] = pk2bf(x1[2], x1[3]);
  r[4] = pk2bf(y0[0], y0[1]); r[5] = pk2bf(y0[2], y0[3]);
  r[6] = pk2bf(y1[0], y1[1]); r[7] = pk2bf(y1[2], y1[3]);
  return r;
}

// Read a 16x32 P tile from LDS into A-fragment layout.
__device__ __forceinline__ u32x8 read_pa(const unsigned short* base, int l15,
                                         int hi) {
  const unsigned short* rowp = base + l15 * 32;
  u32x4 x = *(const u32x4*)(rowp + hi * 8);
  u32x4 y = *(const u32x4*)(rowp + hi * 8 + 16);
  u32x8 r;
  r[0] = x[0]; r[1] = x[1]; r[2] = x[2]; r[3] = x[3];
  r[4] = y[0]; r[5] = y[1]; r[6] = y[2]; r[7] = y[3];
  return r;
}

// ----------------------- TDM / LDS tile staging -----------------------------
// Stage a 128(N) x 32(K) bf16 tile of Bt (row stride = ldk halves) into LDS
// laid out as [128][32] halves.  TDM path: wave 0 issues one tensor op;
// fallback: cooperative b128 copy.  Both produce identical LDS contents.

#define TDM_BIGDIM (1u << 20)   // tensor dims: large so OOB never triggers

__device__ __forceinline__ void stage_b_tile(unsigned short* dst,
                                             const unsigned short* src,
                                             int ldk, int tid) {
#if defined(HAVE_TDM)
  if (tid < 32) {  // one wave issues the DMA (EXEC-independent instruction)
    unsigned lds_off = (unsigned)(unsigned long long)(void*)dst;
    unsigned long long ga = (unsigned long long)src;
    tdm_u32x4 g0;
    g0[0] = 1u;                                    // count=1, user descriptor
    g0[1] = lds_off;                               // lds_addr (bytes)
    g0[2] = (unsigned)ga;                          // global_addr[31:0]
    g0[3] = (unsigned)((ga >> 32) & 0x01FFFFFFu)   // global_addr[56:32]
            | (2u << 30);                          // type = 2 ("image")
    tdm_i32x8 g1;
    g1[0] = 0x00010000;                            // data_size = 1 -> 2 bytes
    g1[1] = (int)((TDM_BIGDIM & 0xFFFFu) << 16);                 // dim0 lo16
    g1[2] = (int)(((TDM_BIGDIM >> 16) & 0xFFFFu)                 // dim0 hi16
                  | ((TDM_BIGDIM & 0xFFFFu) << 16));             // dim1 lo16
    g1[3] = (int)(((TDM_BIGDIM >> 16) & 0xFFFFu)                 // dim1 hi16
                  | (32u << 16));                                // tile_dim0
    g1[4] = 128;                                   // tile_dim1 (tile_dim2=0)
    g1[5] = (int)(unsigned)ldk;                    // tensor_dim0_stride lo32
    g1[6] = 0;                                     // stride0 hi | stride1 lo
    g1[7] = 0;                                     // stride1 hi
    tdm_i32x4 z4 = {0, 0, 0, 0};
#if defined(TDM_ARITY6)
    tdm_i32x8 z8 = {0, 0, 0, 0, 0, 0, 0, 0};
    __builtin_amdgcn_tensor_load_to_lds(g0, g1, z4, z4, z8, 0);
#else
    __builtin_amdgcn_tensor_load_to_lds(g0, g1, z4, z4, 0);
#endif
  }
#else
  for (int ch = tid; ch < (128 * 32) / 8; ch += 256) {
    int row = ch >> 2;
    int part = ch & 3;
    *(u32x4*)(dst + row * 32 + part * 8) =
        *(const u32x4*)(src + (long)row * ldk + part * 8);
  }
#endif
}

__device__ __forceinline__ void stage_wait(int tid) {
#if defined(HAVE_TDM)
  if (tid < 32) {
#if __has_builtin(__builtin_amdgcn_s_wait_tensorcnt)
    __builtin_amdgcn_s_wait_tensorcnt((short)0);
#else
    asm volatile("s_wait_tensorcnt 0x0" ::: "memory");
#endif
  }
#endif
}

// ------------------------ weight convert + transpose ------------------------
// Wt[n*K + k] = bf16(W[k*N + n])   (W is [K x N] row-major f32)

__global__ void convtrans_k(const float* __restrict__ W,
                            unsigned short* __restrict__ Wt, int K, int N) {
  long t = (long)blockIdx.x * blockDim.x + threadIdx.x;
  int n = (int)(t / K);
  int k = (int)(t % K);
  Wt[t] = f2bf(W[(long)k * N + n]);
}

// ------------------------------- GEMM ---------------------------------------
// C[M x N] = (A[M x K] @ Bt^T) + bias, scaled by alpha.   Bt is [N x K] bf16.
// Block: 256 threads = 8 waves (4 M x 2 N), 128(M) x 128(N) tile, 32x64 per
// wave, K-step 32.  B tile double-buffered in LDS via TDM (or coop copy).

template <bool A_F32, bool OUT_BF16, bool TRANS_STORE>
__global__ __launch_bounds__(256) void gemm_wmma_k(
    const void* __restrict__ Ap, const unsigned short* __restrict__ Bt,
    const float* __restrict__ bias, void* __restrict__ Cp, int K, int N,
    float alpha, int trows) {
  const int tid  = threadIdx.x;
  const int lane = tid & 31;
  const int wave = tid >> 5;
  const int l15  = lane & 15;
  const int hi   = (lane >> 4) & 1;
  const long m0  = (long)blockIdx.x * 128 + (wave >> 1) * 32;
  const int  nb  = blockIdx.y * 128;         // block N origin
  const int  nw  = (wave & 1) * 64;          // wave N origin within block

  __shared__ __align__(16) unsigned short btile[2][128][32];

  v8f c[2][4];
#pragma unroll
  for (int mi = 0; mi < 2; ++mi)
#pragma unroll
    for (int ni = 0; ni < 4; ++ni) c[mi][ni] = vzero8();

  stage_b_tile(&btile[0][0][0], Bt + (long)nb * K, K, tid);

  int cur = 0;
  for (int k0 = 0; k0 < K; k0 += 32, cur ^= 1) {
    stage_wait(tid);     // TDM tile `cur` complete (wave 0), then publish
    __syncthreads();

    u32x8 a[2];
#pragma unroll
    for (int mi = 0; mi < 2; ++mi) {
      long row = m0 + mi * 16 + l15;
      int kb = k0 + hi * 8;
      if constexpr (A_F32)
        a[mi] = load_a_f32((const float*)Ap, row, K, kb);
      else
        a[mi] = load_a_bf16((const unsigned short*)Ap, row, K, kb);
    }
    u32x8 bb[4];
#pragma unroll
    for (int ni = 0; ni < 4; ++ni)
      bb[ni] = load_b16(&btile[cur][nw + ni * 16 + l15][hi * 8]);

    if (k0 + 32 < K)  // overlap: DMA next tile into the other buffer
      stage_b_tile(&btile[cur ^ 1][0][0], Bt + (long)nb * K + (k0 + 32), K,
                   tid);

    if (k0 + 32 < K)  // warm caches for the next A tile (global_prefetch_b8)
      __builtin_prefetch((const char*)Ap +
                             ((m0 + l15) * (long)K + k0 + 32) *
                                 (A_F32 ? 4 : 2),
                         0, 0);

#pragma unroll
    for (int mi = 0; mi < 2; ++mi)
#pragma unroll
      for (int ni = 0; ni < 4; ++ni)
        c[mi][ni] = wmma_bf16(a[mi], bb[ni], c[mi][ni]);
  }

#pragma unroll
  for (int mi = 0; mi < 2; ++mi)
#pragma unroll
    for (int ni = 0; ni < 4; ++ni) {
      int coln = nb + nw + ni * 16 + l15;
      float bv = bias[coln];
#pragma unroll
      for (int r = 0; r < 8; ++r) {
        long row = m0 + mi * 16 + r + hi * 8;
        float v = (c[mi][ni][r] + bv) * alpha;
        if constexpr (OUT_BF16) {
          long idx;
          if constexpr (TRANS_STORE)
            idx = ((row / trows) * (long)N + coln) * (long)trows +
                  (row % trows);
          else
            idx = row * (long)N + coln;
          ((unsigned short*)Cp)[idx] = f2bf(v);
        } else {
          ((float*)Cp)[row * (long)N + coln] = v;
        }
      }
    }
}

// ------------------------- attention pass 1 (flash) -------------------------
// Per (b,h) block: q (128x64) vs K/V (4096x64).  Online softmax over s.
// Outputs: tgvT[bh][d][t] = (P@V)[t][d] / l[t]^2  (bf16),  mbuf[bh][t] = m[t].

__global__ __launch_bounds__(256) void attn_pass1(
    const unsigned short* __restrict__ Q, const unsigned short* __restrict__ K2,
    const unsigned short* __restrict__ Vt, unsigned short* __restrict__ tgvT,
    float* __restrict__ mbuf) {
  const int bh = blockIdx.x;
  const int b = bh >> 4, h = bh & 15;
  const int lane = threadIdx.x & 31;
  const int wave = threadIdx.x >> 5;  // 8 waves x 16 t-rows
  const int l15 = lane & 15;
  const int hi = (lane >> 4) & 1;

  __shared__ __align__(16) unsigned short plds[8][16][32];

  // q A-fragments (reused for the whole s loop)
  u32x8 aq[2];
  {
    long qrow = (long)b * LT + wave * 16 + l15;
#pragma unroll
    for (int f = 0; f < 2; ++f)
      aq[f] = load_a_bf16(Q, qrow, DM, h * HD + f * 32 + hi * 8);
  }

  float mr[8], lr[8];
  v8f acc[4];
#pragma unroll
  for (int r = 0; r < 8; ++r) { mr[r] = -1e30f; lr[r] = 0.f; }
#pragma unroll
  for (int n = 0; n < 4; ++n) acc[n] = vzero8();

  for (int s0 = 0; s0 < LV; s0 += 32) {
    // S[j] = q @ k_tile^T for s columns [s0+16j, s0+16j+16)
    v8f S[2];
#pragma unroll
    for (int j = 0; j < 2; ++j) {
      const unsigned short* krow =
          K2 + ((long)b * LV + s0 + j * 16 + l15) * DM + h * HD;
      u32x8 b0 = load_b16(krow + hi * 8);
      u32x8 b1 = load_b16(krow + 32 + hi * 8);
      v8f z = vzero8();
      z = wmma_bf16(aq[0], b0, z);
      z = wmma_bf16(aq[1], b1, z);
      S[j] = z;
    }

    // online softmax (row = t, spread over 16-lane groups)
    float p0[8], p1[8];
#pragma unroll
    for (int r = 0; r < 8; ++r) {
      float tmax = fmaxf(S[0][r], S[1][r]);
      for (int off = 8; off; off >>= 1)
        tmax = fmaxf(tmax, __shfl_xor(tmax, off, 16));
      float nm = fmaxf(mr[r], tmax);
      float fc = __expf(mr[r] - nm);
      float e0 = __expf(S[0][r] - nm);
      float e1 = __expf(S[1][r] - nm);
      float rs = e0 + e1;
      for (int off = 8; off; off >>= 1) rs += __shfl_xor(rs, off, 16);
      lr[r] = lr[r] * fc + rs;
      mr[r] = nm;
      p0[r] = e0; p1[r] = e1;
#pragma unroll
      for (int n = 0; n < 4; ++n) acc[n][r] *= fc;
    }

    // C-layout -> A-layout transpose of P through LDS
#pragma unroll
    for (int r = 0; r < 8; ++r) {
      int row = r + hi * 8;
      plds[wave][row][l15] = f2bf(p0[r]);
      plds[wave][row][16 + l15] = f2bf(p1[r]);
    }
    __syncthreads();
    u32x8 pa = read_pa(&plds[wave][0][0], l15, hi);
    __syncthreads();

    // tgv += P @ V  (B-frags contiguous thanks to transposed V)
#pragma unroll
    for (int n = 0; n < 4; ++n) {
      u32x8 bv = load_b16(Vt + ((long)b * DM + h * HD + n * 16 + l15) * LV +
                          s0 + hi * 8);
      acc[n] = wmma_bf16(pa, bv, acc[n]);
    }
  }

  // tgvT[d][t] = acc[t][d] / l[t]^2
  long tbase = (long)bh * HD * LT;
#pragma unroll
  for (int r = 0; r < 8; ++r) {
    float inv = 1.0f / (lr[r] * lr[r]);
    int t = wave * 16 + r + hi * 8;
#pragma unroll
    for (int n = 0; n < 4; ++n) {
      int d = n * 16 + l15;
      tgvT[tbase + (long)d * LT + t] = f2bf(acc[n][r] * inv);
    }
  }
  if (l15 == 0) {
#pragma unroll
    for (int r = 0; r < 8; ++r)
      mbuf[bh * LT + wave * 16 + r + hi * 8] = mr[r];
  }
}

// ------------------------- attention pass 2 ---------------------------------
// out[s][d] = sum_t exp(S[t][s] - m[t]) * tgvT[d][t]   (per (b,h))
// Block: (bh, s-chunk of 128).  St computed as K_rows x Q^T so s lands on M.

__global__ __launch_bounds__(256) void attn_pass2(
    const unsigned short* __restrict__ Q, const unsigned short* __restrict__ K2,
    const unsigned short* __restrict__ tgvT, const float* __restrict__ mbuf,
    unsigned short* __restrict__ O2d) {
  const int bh = blockIdx.x;
  const int b = bh >> 4, h = bh & 15;
  const int lane = threadIdx.x & 31;
  const int wave = threadIdx.x >> 5;
  const int l15 = lane & 15;
  const int hi = (lane >> 4) & 1;

  __shared__ __align__(16) unsigned short plds[8][16][32];

  const long sbase = (long)b * LV + blockIdx.y * 128 + wave * 16;
  u32x8 ak[2];
#pragma unroll
  for (int f = 0; f < 2; ++f)
    ak[f] = load_a_bf16(K2, sbase + l15, DM, h * HD + f * 32 + hi * 8);

  v8f acc[4];
#pragma unroll
  for (int n = 0; n < 4; ++n) acc[n] = vzero8();

  for (int t0 = 0; t0 < LT; t0 += 32) {
#pragma unroll
    for (int j = 0; j < 2; ++j) {
      const unsigned short* qrow =
          Q + ((long)b * LT + t0 + j * 16 + l15) * DM + h * HD;
      u32x8 b0 = load_b16(qrow + hi * 8);
      u32x8 b1 = load_b16(qrow + 32 + hi * 8);
      v8f z = vzero8();
      z = wmma_bf16(ak[0], b0, z);
      z = wmma_bf16(ak[1], b1, z);
      float mv = mbuf[bh * LT + t0 + j * 16 + l15];  // per-column t
#pragma unroll
      for (int r = 0; r < 8; ++r)
        plds[wave][r + hi * 8][j * 16 + l15] = f2bf(__expf(z[r] - mv));
    }
    __syncthreads();
    u32x8 pa = read_pa(&plds[wave][0][0], l15, hi);
    __syncthreads();

#pragma unroll
    for (int n = 0; n < 4; ++n) {
      u32x8 bt = load_b16(tgvT + (long)bh * HD * LT + (n * 16 + l15) * LT +
                          t0 + hi * 8);
      acc[n] = wmma_bf16(pa, bt, acc[n]);
    }
  }

#pragma unroll
  for (int n = 0; n < 4; ++n) {
    int col = h * HD + n * 16 + l15;
#pragma unroll
    for (int r = 0; r < 8; ++r) {
      long row = sbase + r + hi * 8;
      O2d[row * DM + col] = f2bf(acc[n][r]);
    }
  }
}

// ------------------------------ launcher ------------------------------------

extern "C" void kernel_launch(void* const* d_in, const int* in_sizes, int n_in,
                              void* d_out, int out_size, void* d_ws,
                              size_t ws_size, hipStream_t stream) {
  (void)in_sizes; (void)n_in; (void)out_size; (void)ws_size;

  const float* hidden = (const float*)d_in[0];   // (B, Lvis, D)
  const float* text   = (const float*)d_in[1];   // (Ltext, B, D) -> flat 2048xD
  const float* Wq = (const float*)d_in[2];
  const float* bq = (const float*)d_in[3];
  const float* Wk = (const float*)d_in[4];
  const float* bk = (const float*)d_in[5];
  const float* Wv = (const float*)d_in[6];
  const float* bv = (const float*)d_in[7];
  const float* Wo = (const float*)d_in[8];
  const float* bo = (const float*)d_in[9];

  char* ws = (char*)d_ws;
  size_t off = 0;
  auto alloc = [&](size_t bytes) -> char* {
    char* p = ws + off;
    off = (off + bytes + 255) & ~(size_t)255;
    return p;
  };

  unsigned short* wqT  = (unsigned short*)alloc((size_t)DM * DM * 2);
  unsigned short* wkT  = (unsigned short*)alloc((size_t)DM * DM * 2);
  unsigned short* wvT  = (unsigned short*)alloc((size_t)DM * DM * 2);
  unsigned short* woT  = (unsigned short*)alloc((size_t)DM * DM * 2);
  unsigned short* Q2d  = (unsigned short*)alloc((size_t)NB * LT * DM * 2);
  unsigned short* K2d  = (unsigned short*)alloc((size_t)NB * LV * DM * 2);
  unsigned short* VtT  = (unsigned short*)alloc((size_t)NB * DM * LV * 2);
  unsigned short* tgvT = (unsigned short*)alloc((size_t)NB * NH * HD * LT * 2);
  float*          mbuf = (float*)alloc((size_t)NB * NH * LT * 4);
  unsigned short* O2d  = (unsigned short*)alloc((size_t)NB * LV * DM * 2);

  dim3 blk(256);
  const int wgrid = (DM * DM) / 256;
  convtrans_k<<<wgrid, blk, 0, stream>>>(Wq, wqT, DM, DM);
  convtrans_k<<<wgrid, blk, 0, stream>>>(Wk, wkT, DM, DM);
  convtrans_k<<<wgrid, blk, 0, stream>>>(Wv, wvT, DM, DM);
  convtrans_k<<<wgrid, blk, 0, stream>>>(Wo, woT, DM, DM);

  const float scale = 0.125f;  // hd^-0.5 = 64^-0.5

  // Q = (text @ Wq + bq) * scale   (flat 2048 rows; view trick -> b*128+t)
  gemm_wmma_k<true, true, false>
      <<<dim3((NB * LT) / 128, DM / 128), blk, 0, stream>>>(
          text, wqT, bq, Q2d, DM, DM, scale, 0);
  // K = hidden @ Wk + bk  (token-major)
  gemm_wmma_k<true, true, false>
      <<<dim3((NB * LV) / 128, DM / 128), blk, 0, stream>>>(
          hidden, wkT, bk, K2d, DM, DM, 1.0f, 0);
  // V = hidden @ Wv + bv, stored transposed per batch: Vt[b*DM + c][s]
  gemm_wmma_k<true, true, true>
      <<<dim3((NB * LV) / 128, DM / 128), blk, 0, stream>>>(
          hidden, wvT, bv, VtT, DM, DM, 1.0f, LV);

  attn_pass1<<<dim3(NB * NH), blk, 0, stream>>>(Q2d, K2d, VtT, tgvT, mbuf);
  attn_pass2<<<dim3(NB * NH, LV / 128), blk, 0, stream>>>(Q2d, K2d, tgvT, mbuf,
                                                          O2d);

  // out = O @ Wo + bo  (f32 result)
  gemm_wmma_k<false, false, false>
      <<<dim3((NB * LV) / 128, DM / 128), blk, 0, stream>>>(
          O2d, woT, bo, (float*)d_out, DM, DM, 1.0f, 0);
}
